// DeepECT_695784702595
// MI455X (gfx1250) — compile-verified, matrix operand order checked by reading, not run on hardware
//
#include <hip/hip_runtime.h>
#include <math.h>

// ---------------------------------------------------------------------------
// DeepECT assignment + cosine loss + EMA weights for MI455X (gfx1250).
// Core GEMM (z[262144,64] @ centers^T[64,256]) runs on v_wmma_f32_16x16x32_f16.
// Argmin uses a packed (score|index) key + v_min_u32 to keep the VALU epilogue
// from starving the matrix pipe.
// ---------------------------------------------------------------------------

typedef __attribute__((ext_vector_type(16))) _Float16 v16h;
typedef __attribute__((ext_vector_type(8)))  _Float16 v8h;
typedef __attribute__((ext_vector_type(8)))  float    v8f;
typedef __attribute__((ext_vector_type(4)))  float    f4;

#define NROW 262144
#define KC   256
#define DD   64
#define MT   128          // rows per workgroup (8 waves x 16 rows)
#define LDC  80           // LDS center row stride in halves (160 B, 16B aligned)
#define EPSV 1e-8f
#define BIAS 1024.0f      // makes all scores positive -> uint-orderable bits

// --- kernel 1: exact fp32 ||c||^2 per center + zero the count scratch -------
__global__ __launch_bounds__(256) void deepect_csq_kernel(
    const float* __restrict__ centers, float* __restrict__ csq,
    unsigned int* __restrict__ counts) {
  int k = threadIdx.x;                       // 256 threads, one per center
  const f4* c4 = (const f4*)(centers + k * DD);
  float s = 0.f;
#pragma unroll
  for (int i = 0; i < 16; ++i) {
    f4 v = c4[i];
    s += v[0]*v[0] + v[1]*v[1] + v[2]*v[2] + v[3]*v[3];
  }
  csq[k] = s;
  counts[k] = 0u;
}

// --- kernel 2: WMMA distance GEMM + fused packed-key argmin + cosine --------
__global__ __launch_bounds__(256) void deepect_main_kernel(
    const float* __restrict__ z, const float* __restrict__ centers,
    const float* __restrict__ csq_g, unsigned int* __restrict__ counts_g,
    float* __restrict__ cosd, int* __restrict__ asg) {
  __shared__ _Float16    ldsC[KC * LDC];     // centers as f16, padded rows
  __shared__ float       ldsCsq[KC];
  __shared__ float       ldsZn[MT];
  __shared__ unsigned    ldsCnt[KC];

  const int tid  = threadIdx.x;
  const int wave = tid >> 5;                 // 8 waves of 32
  const int lane = tid & 31;
  const int l16  = lane & 15;
  const int hi   = lane >> 4;                // lane half-group (K sub-layout)
  const int kb   = hi * 8;

  // Stage centers fp32 -> f16 into LDS (coalesced over D).
  for (int idx = tid; idx < KC * DD / 4; idx += 256) {
    int k = idx >> 4;                        // 16 float4s per row
    int d = (idx & 15) * 4;
    f4 v = *(const f4*)(centers + k * DD + d);
    _Float16* dst = &ldsC[k * LDC + d];
    dst[0] = (_Float16)v[0]; dst[1] = (_Float16)v[1];
    dst[2] = (_Float16)v[2]; dst[3] = (_Float16)v[3];
  }
  if (tid < KC) { ldsCsq[tid] = csq_g[tid]; ldsCnt[tid] = 0u; }

  // Build A fragments (16-bit A 16x32 lane layout) and exact fp32 ||z||.
  const int  rowLocal = wave * 16 + l16;
  const long row      = (long)blockIdx.x * MT + rowLocal;
  const float* zr     = z + row * DD;
  v16h a[2];
  float ss = 0.f;
#pragma unroll
  for (int q = 0; q < 2; ++q) {              // K chunks of 32
    f4 p0 = *(const f4*)(zr + q * 32 + kb);
    f4 p1 = *(const f4*)(zr + q * 32 + kb + 4);
    f4 p2 = *(const f4*)(zr + q * 32 + kb + 16);
    f4 p3 = *(const f4*)(zr + q * 32 + kb + 20);
#pragma unroll
    for (int i = 0; i < 4; ++i) {
      float x0 = p0[i], x1 = p1[i], x2 = p2[i], x3 = p3[i];
      ss += x0*x0 + x1*x1 + x2*x2 + x3*x3;
      a[q][i]      = (_Float16)x0;           // K = kb + i
      a[q][4 + i]  = (_Float16)x1;           // K = kb + 4 + i
      a[q][8 + i]  = (_Float16)x2;           // K = kb + 16 + i
      a[q][12 + i] = (_Float16)x3;           // K = kb + 20 + i
    }
  }
  ss += __shfl_xor(ss, 16);                  // lane pair (j, j+16) covers row
  if (hi == 0) ldsZn[rowLocal] = sqrtf(ss);

  __syncthreads();

  // Tile loop over 16 column tiles of 16 centers; packed-key argmin epilogue.
  // key = (bits(biased_score) & ~0xFF) | center_index.
  // Biased score is always positive, so unsigned-int ordering == float
  // ordering; equal quantized scores tie-break to the smaller index (matches
  // argmin-first semantics).
  unsigned bk[8];
#pragma unroll
  for (int e = 0; e < 8; ++e) bk[e] = 0xFFFFFFFFu;

#pragma unroll
  for (int t = 0; t < 16; ++t) {
    const int n = t * 16 + l16;              // this lane's center column
    const _Float16* bp = &ldsC[n * LDC];
    v8f acc = {};
#pragma unroll
    for (int q = 0; q < 2; ++q) {
      v8h blo = *(const v8h*)(bp + q * 32 + kb);        // ds_load_b128
      v8h bhi = *(const v8h*)(bp + q * 32 + kb + 16);   // ds_load_b128
      v16h bb = __builtin_shufflevector(blo, bhi,
                  0,1,2,3,4,5,6,7,8,9,10,11,12,13,14,15);
      acc = __builtin_amdgcn_wmma_f32_16x16x32_f16(
                false, a[q], false, bb, (short)0, acc, false, false);
    }
    const float csb = ldsCsq[n] + BIAS;      // biased ||c||^2
#pragma unroll
    for (int e = 0; e < 8; ++e) {            // D layout: VGPR e -> row hi*8+e
      float    sc  = fmaf(-2.f, acc[e], csb);            // v_fma_f32
      unsigned key = (__float_as_uint(sc) & 0xFFFFFF00u) | (unsigned)n;
      bk[e] = bk[e] < key ? bk[e] : key;                 // v_min_u32
    }
  }

  // Cross-lane argmin over the 16 lanes sharing each row group.
#pragma unroll
  for (int m = 1; m < 16; m <<= 1) {
#pragma unroll
    for (int e = 0; e < 8; ++e) {
      unsigned ok = __shfl_xor(bk[e], m);
      bk[e] = bk[e] < ok ? bk[e] : ok;
    }
  }

  // Write-out: one lane per row emits assignment, cosine loss, LDS count.
  const long rbase = (long)blockIdx.x * MT + wave * 16 + hi * 8;
#pragma unroll
  for (int e = 0; e < 8; ++e) {
    if (l16 == e) {
      const long     g   = rbase + e;
      const unsigned key = bk[e];
      const int      k   = (int)(key & 255u);
      // Recover winning dot from the quantized biased score.
      float scr = __uint_as_float(key & 0xFFFFFF00u);
      float dot = (ldsCsq[k] + BIAS - scr) * 0.5f;
      float zn  = ldsZn[wave * 16 + hi * 8 + e];
      float cn  = sqrtf(ldsCsq[k]);
      zn = zn < EPSV ? EPSV : zn;
      cn = cn < EPSV ? EPSV : cn;
      cosd[g] = 1.f - dot / (zn * cn);
      asg[g]  = k;
      atomicAdd(&ldsCnt[k], 1u);
    }
  }
  __syncthreads();
  if (tid < KC) {
    unsigned c = ldsCnt[tid];
    if (c) atomicAdd(&counts_g[tid], c);
  }
}

// --- kernel 3: EMA weight update -------------------------------------------
__global__ __launch_bounds__(256) void deepect_weight_kernel(
    const float* __restrict__ weights, const unsigned int* __restrict__ counts,
    float* __restrict__ nw) {
  int k = threadIdx.x;
  nw[k] = 0.5f * weights[k] + 0.5f * ((float)counts[k] * (1.f / (float)NROW));
}

extern "C" void kernel_launch(void* const* d_in, const int* in_sizes, int n_in,
                              void* d_out, int out_size, void* d_ws, size_t ws_size,
                              hipStream_t stream) {
  (void)in_sizes; (void)n_in; (void)out_size; (void)ws_size;
  const float* z       = (const float*)d_in[0];   // [N, 64]
  const float* centers = (const float*)d_in[1];   // [256, 64]
  const float* weights = (const float*)d_in[2];   // [256]

  float* out  = (float*)d_out;
  float* cosd = out;                               // [N] cos_dist
  float* nw   = out + NROW;                        // [256] new_weights
  int*   asg  = (int*)(out + NROW + KC);           // [N] assignments (int32)

  float*        csq    = (float*)d_ws;
  unsigned int* counts = (unsigned int*)((char*)d_ws + KC * sizeof(float));

  deepect_csq_kernel<<<1, 256, 0, stream>>>(centers, csq, counts);
  deepect_main_kernel<<<NROW / MT, 256, 0, stream>>>(z, centers, csq, counts,
                                                     cosd, asg);
  deepect_weight_kernel<<<1, 256, 0, stream>>>(weights, counts, nw);
}
